// LlamaAttention_44796508897403
// MI455X (gfx1250) — matrix operand order repaired and checked by reading, not compiled, not run on hardware
//
#include <hip/hip_runtime.h>
#include <math.h>

typedef __bf16 bf16_t;
typedef __attribute__((ext_vector_type(16))) __bf16 v16bf;
typedef __attribute__((ext_vector_type(8)))  __bf16 bf16x8;
typedef __attribute__((ext_vector_type(8)))  float  v8f;
typedef __attribute__((ext_vector_type(4)))  int    v4i;

constexpr int B_     = 4;
constexpr int NEW_   = 512;
constexpr int CACHE_ = 4096;
constexpr int TOTAL_ = CACHE_ + NEW_;
constexpr int H_     = 32;
constexpr int HKV_   = 8;
constexpr int G_     = H_ / HKV_;
constexpr int D_     = 128;
constexpr int HID_   = 4096;
constexpr int BSZ_   = B_ * NEW_;

#define SCALE_QK 0.08838834764831845f  // 1/sqrt(128), folded into Q at RoPE time

// async global->LDS copy (CDNA5), guarded so host pass / older toolchains fall back
#if defined(__gfx1250__) && defined(__has_builtin)
# if __has_builtin(__builtin_amdgcn_global_load_async_to_lds_b128)
#  define HAVE_ASYNC_LDS 1
# endif
#endif
#ifndef HAVE_ASYNC_LDS
# define HAVE_ASYNC_LDS 0
#endif

#if HAVE_ASYNC_LDS
typedef __attribute__((address_space(1))) v4i global_v4i;
typedef __attribute__((address_space(3))) v4i lds_v4i;
#endif

// ---------------- WMMA helpers ----------------

__device__ inline v16bf frag16(const bf16_t* lo, const bf16_t* hi) {
    bf16x8 a = *(const bf16x8*)lo;
    bf16x8 b = *(const bf16x8*)hi;
    v16bf r;
#pragma unroll
    for (int i = 0; i < 8; ++i) { r[i] = a[i]; r[8 + i] = b[i]; }
    return r;
}

__device__ inline v8f wmma_bf16(v16bf a, v16bf b, v8f c) {
    return __builtin_amdgcn_wmma_f32_16x16x32_bf16(
        /*neg_a=*/false, a, /*neg_b=*/false, b,
        /*c_mod=*/(short)0, c, /*reuse_a=*/false, /*reuse_b=*/false);
}

// butterfly reductions within 16-lane halves of a wave32
__device__ inline float half_max(float v) {
#pragma unroll
    for (int m = 1; m <= 8; m <<= 1) v = fmaxf(v, __shfl_xor(v, m, 32));
    return v;
}
__device__ inline float half_sum(float v) {
#pragma unroll
    for (int m = 1; m <= 8; m <<= 1) v += __shfl_xor(v, m, 32);
    return v;
}

// ---------------- RMSNorm ----------------

__global__ void rmsnorm_k(const float* __restrict__ x, const float* __restrict__ w,
                          bf16_t* __restrict__ hn) {
    const int row = blockIdx.x, t = threadIdx.x;
    const float* xr = x + (size_t)row * HID_;
    float ss = 0.f;
    for (int i = t; i < HID_; i += 256) { float v = xr[i]; ss += v * v; }
    __shared__ float red[256];
    red[t] = ss; __syncthreads();
    for (int s = 128; s > 0; s >>= 1) { if (t < s) red[t] += red[t + s]; __syncthreads(); }
    const float rinv = rsqrtf(red[0] * (1.0f / HID_) + 1e-6f);
    bf16_t* out = hn + (size_t)row * HID_;
    for (int i = t; i < HID_; i += 256) out[i] = (bf16_t)(w[i] * xr[i] * rinv);
}

// ---------------- fp32 -> bf16 convert ----------------

__global__ void cvt_k(const float* __restrict__ in, bf16_t* __restrict__ out, long n) {
    long i = (long)blockIdx.x * blockDim.x + threadIdx.x;
    const long stride = (long)gridDim.x * blockDim.x;
    for (; i < n; i += stride) out[i] = (bf16_t)in[i];
}

// ---------------- KV cache bf16 conversion (old part) ----------------

__global__ void cachecvt_k(const float* __restrict__ kin, const float* __restrict__ vin,
                           bf16_t* __restrict__ kc, bf16_t* __restrict__ vc) {
    const long n = (long)B_ * CACHE_ * HKV_ * D_;
    long i = (long)blockIdx.x * blockDim.x + threadIdx.x;
    const long stride = (long)gridDim.x * blockDim.x;
    for (; i < n; i += stride) {
        long b   = i / ((long)CACHE_ * HKV_ * D_);
        long rem = i - b * ((long)CACHE_ * HKV_ * D_);
        long t   = rem / (HKV_ * D_);
        long rr  = rem - t * (HKV_ * D_);
        long off = (b * TOTAL_ + t) * (long)(HKV_ * D_) + rr;
        kc[off] = (bf16_t)kin[off];
        vc[off] = (bf16_t)vin[off];
    }
}

// ---------------- RoPE on Q (in place, bf16); folds softmax scale ----------------

__global__ void rope_q_k(bf16_t* __restrict__ q, const float* __restrict__ cosv,
                         const float* __restrict__ sinv) {
    const long n = (long)BSZ_ * H_ * (D_ / 2);
    long i = (long)blockIdx.x * blockDim.x + threadIdx.x;
    if (i >= n) return;
    long t = i / (H_ * (D_ / 2));
    long r = i - t * (H_ * (D_ / 2));
    int  h = (int)(r / (D_ / 2));
    int  d = (int)(r - (long)h * (D_ / 2));
    bf16_t* qp = q + (t * H_ + h) * (long)D_;
    float x1 = (float)qp[d], x2 = (float)qp[d + D_ / 2];
    float c1 = cosv[t * D_ + d],          s1 = sinv[t * D_ + d];
    float c2 = cosv[t * D_ + d + D_ / 2], s2 = sinv[t * D_ + d + D_ / 2];
    qp[d]          = (bf16_t)((x1 * c1 - x2 * s1) * SCALE_QK);
    qp[d + D_ / 2] = (bf16_t)((x2 * c2 + x1 * s2) * SCALE_QK);
}

// ---------------- RoPE on new K + scatter into bf16 cache ----------------

__global__ void rope_k_scatter_k(const bf16_t* __restrict__ knew, const float* __restrict__ cosv,
                                 const float* __restrict__ sinv, bf16_t* __restrict__ kc) {
    const long n = (long)BSZ_ * HKV_ * (D_ / 2);
    long i = (long)blockIdx.x * blockDim.x + threadIdx.x;
    if (i >= n) return;
    long t = i / (HKV_ * (D_ / 2));
    long r = i - t * (HKV_ * (D_ / 2));
    int  h = (int)(r / (D_ / 2));
    int  d = (int)(r - (long)h * (D_ / 2));
    const bf16_t* kp = knew + (t * HKV_ + h) * (long)D_;
    float x1 = (float)kp[d], x2 = (float)kp[d + D_ / 2];
    float c1 = cosv[t * D_ + d],          s1 = sinv[t * D_ + d];
    float c2 = cosv[t * D_ + d + D_ / 2], s2 = sinv[t * D_ + d + D_ / 2];
    long b = t / NEW_, tok = t - b * NEW_;
    bf16_t* out = kc + ((b * TOTAL_ + CACHE_ + tok) * HKV_ + h) * (long)D_;
    out[d]          = (bf16_t)(x1 * c1 - x2 * s1);
    out[d + D_ / 2] = (bf16_t)(x2 * c2 + x1 * s2);
}

__global__ void v_scatter_k(const bf16_t* __restrict__ vnew, bf16_t* __restrict__ vc) {
    const long n = (long)BSZ_ * HKV_ * D_;
    long i = (long)blockIdx.x * blockDim.x + threadIdx.x;
    if (i >= n) return;
    long t  = i / (HKV_ * D_);
    long rr = i - t * (HKV_ * D_);
    long b = t / NEW_, tok = t - b * NEW_;
    vc[(b * TOTAL_ + CACHE_ + tok) * (long)(HKV_ * D_) + rr] = vnew[i];
}

// ---------------- WMMA GEMM:  C[MxN] = A[MxK] @ B[KxN]  (bf16 in) ----------------
// block tile 64x128, 8 waves; wave = 16x64 strip; K step 32.

template <bool F32OUT>
__global__ __launch_bounds__(256, 1)
void gemm_wmma_k(const bf16_t* __restrict__ A, const bf16_t* __restrict__ Bm,
                 bf16_t* __restrict__ Cb, float* __restrict__ Cf,
                 const float* __restrict__ res, int M, int N, int K) {
    __shared__ __attribute__((aligned(16))) bf16_t As[64][32];
    __shared__ __attribute__((aligned(16))) bf16_t Bs[128][32];   // [n][k] (transposed)
    const int t = threadIdx.x;
    const int lane = t & 31, w = t >> 5;
    const int n0 = blockIdx.x * 128, m0 = blockIdx.y * 64;
    const int mt = (w & 3) * 16, nb = (w >> 2) * 64;
    const int ln = lane & 15;
    const int boff = (lane < 16) ? 0 : 8;    // A-fragment chunk select
    const int koff = (lane < 16) ? 0 : 16;   // unused; B fragment uses boff as well
    v8f acc[4] = {};
    const int arow = t >> 2, achk = (t & 3) * 8;

    for (int kt = 0; kt < K; kt += 32) {
        // stage A tile (row major)
        *(bf16x8*)&As[arow][achk] =
            *(const bf16x8*)&A[(size_t)(m0 + arow) * K + kt + achk];
        __builtin_prefetch(&A[(size_t)(m0 + arow) * K + kt + 32 + achk], 0, 1);
        // stage B tile transposed: Bs[n][k]
#pragma unroll
        for (int i = 0; i < 2; ++i) {
            int c  = t + i * 256;
            int kr = c >> 4, nc = (c & 15) * 8;
            bf16x8 v = *(const bf16x8*)&Bm[(size_t)(kt + kr) * N + n0 + nc];
#pragma unroll
            for (int j = 0; j < 8; ++j) Bs[nc + j][kr] = v[j];
        }
        __syncthreads();
        v16bf af = frag16(&As[mt + ln][boff], &As[mt + ln][boff + 16]);
#pragma unroll
        for (int j = 0; j < 4; ++j) {
            const bf16_t* cp = &Bs[nb + j * 16 + ln][boff];
            acc[j] = wmma_bf16(af, frag16(cp, cp + 16), acc[j]);
        }
        __syncthreads();
    }

    const int half = lane >> 4;
#pragma unroll
    for (int j = 0; j < 4; ++j)
#pragma unroll
        for (int r = 0; r < 8; ++r) {
            int m = m0 + mt + r + 8 * half;
            int n = n0 + nb + j * 16 + ln;
            size_t off = (size_t)m * N + n;
            if (F32OUT) Cf[off] = acc[j][r] + res[off];
            else        Cb[off] = (bf16_t)acc[j][r];
        }
}

// ---------------- GQA flash attention ----------------
// grid: (NEW/16, HKV, B); block: 128 threads = 4 waves = 4 q-heads of one kv group.

__global__ __launch_bounds__(128, 1)
void attn_k(const bf16_t* __restrict__ q, const bf16_t* __restrict__ kc,
            const bf16_t* __restrict__ vc, bf16_t* __restrict__ out) {
    __shared__ __attribute__((aligned(16))) bf16_t Ks[32][128];     // [key][d]
    __shared__ __attribute__((aligned(16))) bf16_t Vs[128][32];     // [d][key]
    __shared__ __attribute__((aligned(16))) bf16_t Ps[4][16][32];   // per wave [m][key]
    const int qt = blockIdx.x, kvh = blockIdx.y, b = blockIdx.z;
    const int t = threadIdx.x, lane = t & 31, w = t >> 5;
    const int h = kvh * G_ + w;
    const int ln = lane & 15, half = lane >> 4;
    const int boff = half ? 8 : 0, koff = half ? 16 : 0;

    // Q fragments for this wave's 16x128 query tile (A-layout); scale pre-folded
    v16bf aq[4];
    {
        const bf16_t* qp = q + ((size_t)(b * NEW_ + qt * 16 + ln) * H_ + h) * D_;
#pragma unroll
        for (int kd = 0; kd < 4; ++kd)
            aq[kd] = frag16(qp + kd * 32 + boff, qp + kd * 32 + boff + 16);
    }

    v8f acc[8] = {};
    float mrun[8], lrun[8];
#pragma unroll
    for (int r = 0; r < 8; ++r) { mrun[r] = -INFINITY; lrun[r] = 0.f; }

    const int qbase = CACHE_ + qt * 16;
    const int nkb = (qbase + 16 + 31) >> 5;  // key blocks of 32

    for (int kb = 0; kb < nkb; ++kb) {
        const int s0 = kb * 32;
        __syncthreads();
        // stage K (async DMA to LDS when available) and V (transposed, manual)
#pragma unroll
        for (int i = 0; i < 4; ++i) {
            int c = t + i * 128;                 // 0..511
            int key = c >> 4, dc = (c & 15) * 8;
            size_t goff = (((size_t)b * TOTAL_ + s0 + key) * HKV_ + kvh) * D_ + dc;
#if HAVE_ASYNC_LDS
            __builtin_amdgcn_global_load_async_to_lds_b128(
                (global_v4i*)(uintptr_t)&kc[goff],
                (lds_v4i*)&Ks[key][dc], 0, 0);
#else
            *(bf16x8*)&Ks[key][dc] = *(const bf16x8*)&kc[goff];
#endif
            bf16x8 vv = *(const bf16x8*)&vc[goff];
#pragma unroll
            for (int j = 0; j < 8; ++j) Vs[dc + j][key] = vv[j];
        }
#if HAVE_ASYNC_LDS
        asm volatile("s_wait_asynccnt 0" ::: "memory");
#endif
        __syncthreads();

        // scores: S[16x32] = Q @ K^T  (two 16-key WMMA tiles, 4 K-chunks each)
        v8f s0t = {}, s1t = {};
#pragma unroll
        for (int kd = 0; kd < 4; ++kd) {
            const bf16_t* b0 = &Ks[ln][kd * 32 + boff];
            const bf16_t* b1 = &Ks[16 + ln][kd * 32 + boff];
            s0t = wmma_bf16(aq[kd], frag16(b0, b0 + 16), s0t);
            s1t = wmma_bf16(aq[kd], frag16(b1, b1 + 16), s1t);
        }

        // causal mask + online softmax; write P tile (bf16) to per-wave LDS
#pragma unroll
        for (int r = 0; r < 8; ++r) {
            int m = r + 8 * half;
            int qpos = qbase + m;
            float v0 = (s0 + ln      <= qpos) ? s0t[r] : -1e30f;
            float v1 = (s0 + 16 + ln <= qpos) ? s1t[r] : -1e30f;
            float rm = half_max(fmaxf(v0, v1));
            float mn = fmaxf(mrun[r], rm);
            float p0 = __expf(v0 - mn), p1 = __expf(v1 - mn);
            float rs = half_sum(p0 + p1);
            float alpha = __expf(mrun[r] - mn);
            mrun[r] = mn;
            lrun[r] = lrun[r] * alpha + rs;
#pragma unroll
            for (int dt = 0; dt < 8; ++dt) acc[dt][r] *= alpha;
            Ps[w][m][ln]      = (bf16_t)p0;
            Ps[w][m][16 + ln] = (bf16_t)p1;
        }
        // same-wave LDS RAW fence (CDNA5 split counter)
        asm volatile("s_wait_dscnt 0" ::: "memory");

        // P A-fragment, then O += P @ V over 8 d-tiles
        const bf16_t* pp = &Ps[w][ln][0];
        v16bf ap = frag16(pp + boff, pp + boff + 16);
#pragma unroll
        for (int dt = 0; dt < 8; ++dt) {
            const bf16_t* vp = &Vs[dt * 16 + ln][boff];
            acc[dt] = wmma_bf16(ap, frag16(vp, vp + 16), acc[dt]);
        }
    }

    // normalize + store attention output (bf16, [token][h*D+d])
#pragma unroll
    for (int r = 0; r < 8; ++r) {
        int m = r + 8 * half;
        float inv = 1.f / lrun[r];
        size_t base = ((size_t)(b * NEW_ + qt * 16 + m) * H_ + h) * D_;
#pragma unroll
        for (int dt = 0; dt < 8; ++dt)
            out[base + dt * 16 + ln] = (bf16_t)(acc[dt][r] * inv);
    }
}

// ---------------- host launch ----------------

extern "C" void kernel_launch(void* const* d_in, const int* in_sizes, int n_in,
                              void* d_out, int out_size, void* d_ws, size_t ws_size,
                              hipStream_t stream) {
    const float* hidden = (const float*)d_in[0];
    const float* cosv   = (const float*)d_in[1];
    const float* sinv   = (const float*)d_in[2];
    const float* lnw    = (const float*)d_in[3];
    const float* wq     = (const float*)d_in[4];
    const float* wk     = (const float*)d_in[5];
    const float* wv     = (const float*)d_in[6];
    const float* wo     = (const float*)d_in[7];
    const float* kcache = (const float*)d_in[8];
    const float* vcache = (const float*)d_in[9];
    float* outp = (float*)d_out;

    size_t off = 0;
    auto alloc = [&](size_t bytes) {
        size_t r = off;
        off = (off + bytes + 255) & ~(size_t)255;
        return (char*)d_ws + r;
    };
    bf16_t* hn    = (bf16_t*)alloc((size_t)BSZ_ * HID_ * 2);
    bf16_t* wq_b  = (bf16_t*)alloc((size_t)HID_ * H_ * D_ * 2);
    bf16_t* wk_b  = (bf16_t*)alloc((size_t)HID_ * HKV_ * D_ * 2);
    bf16_t* wv_b  = (bf16_t*)alloc((size_t)HID_ * HKV_ * D_ * 2);
    bf16_t* wo_b  = (bf16_t*)alloc((size_t)H_ * D_ * HID_ * 2);
    bf16_t* qbuf  = (bf16_t*)alloc((size_t)BSZ_ * H_ * D_ * 2);
    bf16_t* knew  = (bf16_t*)alloc((size_t)BSZ_ * HKV_ * D_ * 2);
    bf16_t* vnew  = (bf16_t*)alloc((size_t)BSZ_ * HKV_ * D_ * 2);
    bf16_t* kc    = (bf16_t*)alloc((size_t)B_ * TOTAL_ * HKV_ * D_ * 2);
    bf16_t* vc    = (bf16_t*)alloc((size_t)B_ * TOTAL_ * HKV_ * D_ * 2);
    bf16_t* attn  = (bf16_t*)alloc((size_t)BSZ_ * H_ * D_ * 2);
    (void)ws_size; (void)in_sizes; (void)n_in; (void)out_size;

    // 1) RMSNorm -> bf16
    rmsnorm_k<<<BSZ_, 256, 0, stream>>>(hidden, lnw, hn);

    // 2) weight conversions
    cvt_k<<<2048, 256, 0, stream>>>(wq, wq_b, (long)HID_ * H_ * D_);
    cvt_k<<<2048, 256, 0, stream>>>(wk, wk_b, (long)HID_ * HKV_ * D_);
    cvt_k<<<2048, 256, 0, stream>>>(wv, wv_b, (long)HID_ * HKV_ * D_);
    cvt_k<<<2048, 256, 0, stream>>>(wo, wo_b, (long)H_ * D_ * HID_);

    // 3) QKV projections (WMMA, bf16 out)
    gemm_wmma_k<false><<<dim3((H_ * D_) / 128, BSZ_ / 64), 256, 0, stream>>>(
        hn, wq_b, qbuf, nullptr, nullptr, BSZ_, H_ * D_, HID_);
    gemm_wmma_k<false><<<dim3((HKV_ * D_) / 128, BSZ_ / 64), 256, 0, stream>>>(
        hn, wk_b, knew, nullptr, nullptr, BSZ_, HKV_ * D_, HID_);
    gemm_wmma_k<false><<<dim3((HKV_ * D_) / 128, BSZ_ / 64), 256, 0, stream>>>(
        hn, wv_b, vnew, nullptr, nullptr, BSZ_, HKV_ * D_, HID_);

    // 4) bf16 KV cache: old part converted, new part RoPE'd/scattered
    cachecvt_k<<<4096, 256, 0, stream>>>(kcache, vcache, kc, vc);
    {
        long nq = (long)BSZ_ * H_ * (D_ / 2);
        rope_q_k<<<(int)((nq + 255) / 256), 256, 0, stream>>>(qbuf, cosv, sinv);
        long nk = (long)BSZ_ * HKV_ * (D_ / 2);
        rope_k_scatter_k<<<(int)((nk + 255) / 256), 256, 0, stream>>>(knew, cosv, sinv, kc);
        long nv = (long)BSZ_ * HKV_ * D_;
        v_scatter_k<<<(int)((nv + 255) / 256), 256, 0, stream>>>(vnew, vc);
    }

    // 5) attention
    attn_k<<<dim3(NEW_ / 16, HKV_, B_), 128, 0, stream>>>(qbuf, kc, vc, attn);

    // 6) output projection + residual (f32 out)
    gemm_wmma_k<true><<<dim3(HID_ / 128, BSZ_ / 64), 256, 0, stream>>>(
        attn, wo_b, nullptr, outp, hidden, BSZ_, HID_, H_ * D_);
}
